// EdgeDensePredictionGraphScoreNetwork_14422500180385
// MI455X (gfx1250) — matrix operand order repaired and checked by reading, compiled, not run on hardware
//
#include <hip/hip_runtime.h>

// ---------------- types ----------------
typedef __attribute__((ext_vector_type(8)))  __bf16 v8bf;
typedef __attribute__((ext_vector_type(16))) __bf16 v16bf;
typedef __attribute__((ext_vector_type(8)))  float  v8f;
typedef unsigned short u16;

#define BB 16
#define NN 64
#define F_IN 128
#define F_HID 256
#define OUT_DIM 768
#define H_T 1536
#define ROWS 65536          // B*N*N
#define KR_PAD 800          // 770 -> 800  (K multiple of 32)
#define NR_PAD 1664         // 1540 -> 1664 (N multiple of 128)
#define BN_EPS 1e-5f

__device__ __forceinline__ u16 f2bf(float f) {
    union { float f; unsigned u; } v; v.f = f;
    unsigned u = v.u;
    unsigned r = u + 0x7FFFu + ((u >> 16) & 1u);
    return (u16)(r >> 16);
}
__device__ __forceinline__ float bf2f(u16 h) {
    union { unsigned u; float f; } v; v.u = ((unsigned)h) << 16;
    return v.f;
}
__device__ __forceinline__ void store_out(float* p, float v) { *p = v; }
__device__ __forceinline__ void store_out(u16* p, float v) { *p = f2bf(v); }

// CDNA5 async global->LDS 16-byte copy (ASYNCcnt-tracked DMA, bypasses VGPRs)
__device__ __forceinline__ void async_cp16(unsigned lds_off, const void* gaddr) {
    asm volatile("global_load_async_to_lds_b128 %0, %1, off"
                 :: "v"(lds_off), "v"(gaddr) : "memory");
}
__device__ __forceinline__ void wait_async4() {
    asm volatile("s_wait_asynccnt 0x4" ::: "memory");
}
__device__ __forceinline__ void wait_async0() {
    asm volatile("s_wait_asynccnt 0x0" ::: "memory");
}

// ---------------- generic bf16 WMMA GEMM ----------------
// C[M,N] = act(A[M,K] @ Bt[N,K]^T + bias).  A/Bt bf16 row-major.
// Requirements: K % 32 == 0; A has >= ceil128(M) rows, Bt has >= ceil128(N)
// rows (zero-padded) so staging needs no bounds checks. Epilogue guards M/N.
// 256 threads, tile 128x128; 8 waves each compute 64x32 (4x2 wmma tiles).
// Double-buffered LDS fed by global_load_async_to_lds_b128.
template<int ACT, typename OutT>
__global__ __launch_bounds__(256)
void gemm_bf16_wmma(const u16* __restrict__ A,
                    const u16* __restrict__ Bt,
                    const float* __restrict__ bias,
                    OutT* __restrict__ C,
                    int M, int N, int K, int ldc,
                    long sA, long sB, long sC)
{
    A  += (long)blockIdx.z * sA;
    Bt += (long)blockIdx.z * sB;
    C  += (long)blockIdx.z * sC;
    const int m0 = blockIdx.y * 128;
    const int n0 = blockIdx.x * 128;

    __shared__ __align__(16) u16 As[2][128 * 32];
    __shared__ __align__(16) u16 Bs[2][128 * 32];

    const int tid  = threadIdx.x;
    const int lane = tid & 31;
    const int wid  = tid >> 5;
    const int wm   = (wid & 1) * 64;   // wave M offset inside tile
    const int wn   = (wid >> 1) * 32;  // wave N offset inside tile
    const int half = lane & 15;
    const int sel  = lane >> 4;

    // staging: thread handles 16B chunks tid and tid+256 of the 8KB tile
    const int r0 = tid >> 2, s0 = tid & 3;
    const u16* gA0 = A  + (size_t)(m0 + r0) * K + s0 * 8;
    const u16* gA1 = A  + (size_t)(m0 + r0 + 64) * K + s0 * 8;
    const u16* gB0 = Bt + (size_t)(n0 + r0) * K + s0 * 8;
    const u16* gB1 = Bt + (size_t)(n0 + r0 + 64) * K + s0 * 8;
    const unsigned ldsA = ((unsigned)(size_t)&As[0][0]) + tid * 16;
    const unsigned ldsB = ((unsigned)(size_t)&Bs[0][0]) + tid * 16;
    const unsigned BUFB = 128 * 32 * 2;   // bytes per LDS buffer

    // prologue: issue tile 0 into buffer 0
    async_cp16(ldsA, gA0);
    async_cp16(ldsA + 256 * 16, gA1);
    async_cp16(ldsB, gB0);
    async_cp16(ldsB + 256 * 16, gB1);

    v8f acc[4][2];
    for (int mf = 0; mf < 4; ++mf)
        for (int nf = 0; nf < 2; ++nf) {
            int gn = n0 + wn + nf * 16 + half;
            float bv = (bias != nullptr && gn < N) ? bias[gn] : 0.0f;
            for (int r = 0; r < 8; ++r) acc[mf][nf][r] = bv;
        }

    const int nIter = K >> 5;
    for (int it = 0; it < nIter; ++it) {
        const int cur = it & 1;
        if (it + 1 < nIter) {
            // issue next tile into the other buffer (readers of it finished
            // at the barrier that ended the previous iteration)
            gA0 += 32; gA1 += 32; gB0 += 32; gB1 += 32;
            const unsigned o = (cur ^ 1) * BUFB;
            async_cp16(ldsA + o, gA0);
            async_cp16(ldsA + o + 256 * 16, gA1);
            async_cp16(ldsB + o, gB0);
            async_cp16(ldsB + o + 256 * 16, gB1);
            wait_async4();   // oldest 4 (current tile) complete, in-order
        } else {
            wait_async0();
        }
        __syncthreads();

        const u16* curA = &As[cur][0];
        const u16* curB = &Bs[cur][0];
        v16bf af[4], bfr[2];
        for (int mf = 0; mf < 4; ++mf) {
            const u16* p = curA + (wm + mf * 16 + half) * 32;
            v8bf lo = *reinterpret_cast<const v8bf*>(p + sel * 8);        // K: sel*8 .. +7
            v8bf hi = *reinterpret_cast<const v8bf*>(p + 16 + sel * 8);   // K: 16+sel*8 .. +7
            af[mf] = __builtin_shufflevector(lo, hi, 0,1,2,3,4,5,6,7,8,9,10,11,12,13,14,15);
        }
        for (int nf = 0; nf < 2; ++nf) {
            const u16* p = curB + (wn + nf * 16 + half) * 32;
            v8bf lo = *reinterpret_cast<const v8bf*>(p + sel * 8);        // K: sel*8 .. +7
            v8bf hi = *reinterpret_cast<const v8bf*>(p + 16 + sel * 8);   // K: 16+sel*8 .. +7
            bfr[nf] = __builtin_shufflevector(lo, hi, 0,1,2,3,4,5,6,7,8,9,10,11,12,13,14,15);
        }
        for (int mf = 0; mf < 4; ++mf)
            for (int nf = 0; nf < 2; ++nf)
                acc[mf][nf] = __builtin_amdgcn_wmma_f32_16x16x32_bf16(
                    false, af[mf], false, bfr[nf], (short)0, acc[mf][nf], false, false);
        __syncthreads();   // reads done before this buffer is refilled
    }

    for (int mf = 0; mf < 4; ++mf)
        for (int nf = 0; nf < 2; ++nf) {
            int gn = n0 + wn + nf * 16 + half;
            if (gn >= N) continue;
            for (int r = 0; r < 8; ++r) {
                int gm = m0 + wm + mf * 16 + sel * 8 + r;
                if (gm >= M) continue;
                float v = acc[mf][nf][r];
                if (ACT == 1) v = v > 0.0f ? v : (__expf(v) - 1.0f);   // elu
                else if (ACT == 2) v = tanhf(v);
                store_out(&C[(size_t)gm * ldc + gn], v);
            }
        }
}

// ---------------- elementwise / prep kernels ----------------

// transpose+convert weight: dst[n*Kp+k] = src[(rowOff+k)*ld + n] (zero-padded)
__global__ void k_tconv(const float* __restrict__ src, int ld, int rowOff,
                        int Kuse, int Nuse, int Kp, u16* __restrict__ dst,
                        int total)
{
    int idx = blockIdx.x * 256 + threadIdx.x;
    if (idx >= total) return;
    int k = idx % Kp, n = idx / Kp;
    float v = (k < Kuse && n < Nuse) ? src[(size_t)(rowOff + k) * ld + n] : 0.0f;
    dst[idx] = f2bf(v);
}

// masked 2-channel adjacency, bf16, layout [b][c][128][64] (rows 64..127 zero)
__global__ void k_buildA(const float* __restrict__ adjs, const float* __restrict__ flags,
                         u16* __restrict__ Abf)
{
    int idx = blockIdx.x * 256 + threadIdx.x;   // 16*2*128*64
    if (idx >= BB * 2 * 128 * 64) return;
    int j = idx & 63, i = (idx >> 6) & 127, ch = (idx >> 13) & 1, b = idx >> 14;
    float v = 0.0f;
    if (i < 64) {
        float m = flags[b * NN + i] * flags[b * NN + j];
        float a = adjs[b * 4096 + i * 64 + j];
        v = (ch ? (1.0f - a) : a) * m;
    }
    Abf[idx] = f2bf(v);
}

// x -> xh[:,0:128] (bf16) and xT[b][f][j] (bf16, B operand of A@x)
__global__ void k_xprep(const float* __restrict__ x, u16* __restrict__ xh,
                        u16* __restrict__ xT)
{
    int idx = blockIdx.x * 256 + threadIdx.x;   // 16*64*128
    if (idx >= BB * NN * F_IN) return;
    int f = idx & 127, rest = idx >> 7;
    int j = rest & 63, b = rest >> 6;
    u16 v = f2bf(x[idx]);
    xh[(size_t)(b * NN + j) * 384 + f] = v;
    xT[(size_t)b * 8192 + f * 64 + j] = v;
}

// mask h part of xh (cols 128..383) by node flags
__global__ void k_maskh(u16* __restrict__ xh, const float* __restrict__ flags)
{
    int idx = blockIdx.x * 256 + threadIdx.x;   // 1024*256
    if (idx >= BB * NN * F_HID) return;
    int row = idx >> 8, c = idx & 255;
    u16* p = xh + (size_t)row * 384 + 128 + c;
    *p = f2bf(bf2f(*p) * flags[row]);
}

// x_o: apply flag, write f32 to d_out and bf16 copy
__global__ void k_xofin(const float* __restrict__ xo, const float* __restrict__ flags,
                        float* __restrict__ out, u16* __restrict__ xobf)
{
    int idx = blockIdx.x * 256 + threadIdx.x;   // 1024*768
    if (idx >= BB * NN * OUT_DIM) return;
    int row = idx / OUT_DIM;
    float v = xo[idx] * flags[row];
    out[idx] = v;
    xobf[idx] = f2bf(v);
}

// BN statistics over reconstructed hmid_pre (two-pass, nothing materialized)
__global__ __launch_bounds__(256)
void k_ewstat(const float* __restrict__ P, const float* __restrict__ Q,
              const float* __restrict__ adjs, const float* __restrict__ flags,
              const float* __restrict__ w0, const float* __restrict__ w1,
              const float* __restrict__ b1t,
              float* __restrict__ gsum, float* __restrict__ gsq)
{
    int c = blockIdx.y * 256 + threadIdx.x;     // 6 x 256 = 1536
    int rbase = blockIdx.x * 64;                // 1024 row blocks
    float w0c = w0[c], w1c = w1[c], bc = b1t[c];
    float s = 0.0f, q = 0.0f;
    for (int rr = 0; rr < 64; ++rr) {
        int r = rbase + rr;
        int b = r >> 12;
        int rowP = r >> 6;
        int rowQ = (b << 6) | (r & 63);
        float m = flags[rowP] * flags[rowQ];
        float a = adjs[r];
        float h = P[(size_t)rowP * H_T + c] + Q[(size_t)rowQ * H_T + c]
                + (a * m) * w0c + ((1.0f - a) * m) * w1c + bc;
        s += h; q += h * h;
    }
    atomicAdd(&gsum[c], s);
    atomicAdd(&gsq[c], q);
}

__global__ void k_bnfin(const float* __restrict__ gsum, const float* __restrict__ gsq,
                        const float* __restrict__ gamma, const float* __restrict__ beta,
                        float* __restrict__ scale, float* __restrict__ shift)
{
    int c = blockIdx.x * 256 + threadIdx.x;
    if (c >= H_T) return;
    float mean = gsum[c] * (1.0f / ROWS);
    float var  = gsq[c] * (1.0f / ROWS) - mean * mean;
    float sc = gamma[c] * rsqrtf(var + BN_EPS);
    scale[c] = sc;
    shift[c] = beta[c] - mean * sc;
}

// hmid = bf16(elu(BN(hmid_pre)))
__global__ __launch_bounds__(256)
void k_ew2(const float* __restrict__ P, const float* __restrict__ Q,
           const float* __restrict__ adjs, const float* __restrict__ flags,
           const float* __restrict__ w0, const float* __restrict__ w1,
           const float* __restrict__ b1t,
           const float* __restrict__ scale, const float* __restrict__ shift,
           u16* __restrict__ hmid)
{
    int c = blockIdx.y * 256 + threadIdx.x;
    int rbase = blockIdx.x * 64;
    float w0c = w0[c], w1c = w1[c], bc = b1t[c];
    float sc = scale[c], sh = shift[c];
    for (int rr = 0; rr < 64; ++rr) {
        int r = rbase + rr;
        int b = r >> 12;
        int rowP = r >> 6;
        int rowQ = (b << 6) | (r & 63);
        float m = flags[rowP] * flags[rowQ];
        float a = adjs[r];
        float h = P[(size_t)rowP * H_T + c] + Q[(size_t)rowQ * H_T + c]
                + (a * m) * w0c + ((1.0f - a) * m) * w1c + bc;
        float v = h * sc + sh;
        v = v > 0.0f ? v : (__expf(v) - 1.0f);
        hmid[(size_t)r * H_T + c] = f2bf(v);
    }
}

// r_in[r] = [A0, A1, (raw[b,i,j]+raw[b,j,i])*mask (768), zeros(30)]  (bf16)
__global__ __launch_bounds__(256)
void k_rin(const u16* __restrict__ raw, const float* __restrict__ adjs,
           const float* __restrict__ flags, u16* __restrict__ rin)
{
    int r = blockIdx.x;
    int b = r >> 12, ij = r & 4095, i = ij >> 6, j = ij & 63;
    float m = flags[b * NN + i] * flags[b * NN + j];
    float a = adjs[r];
    int rT = (b << 12) | (j << 6) | i;
    for (int c = threadIdx.x; c < KR_PAD; c += 256) {
        float v;
        if (c == 0)       v = a * m;
        else if (c == 1)  v = (1.0f - a) * m;
        else if (c < 770) {
            int cc = c - 2;
            v = (bf2f(raw[(size_t)r * OUT_DIM + cc]) + bf2f(raw[(size_t)rT * OUT_DIM + cc])) * m;
        } else v = 0.0f;
        rin[(size_t)r * KR_PAD + c] = f2bf(v);
    }
}

__global__ void k_padbias(const float* __restrict__ b1r, float* __restrict__ dst)
{
    int i = blockIdx.x * 256 + threadIdx.x;
    if (i >= NR_PAD) return;
    dst[i] = (i < 1540) ? b1r[i] : 0.0f;
}

// score[r] = (i!=j) * (sum_k hr[r,k]*W2r[k] + b2r); one wave per row
__global__ __launch_bounds__(256)
void k_score(const u16* __restrict__ hr, const float* __restrict__ W2r,
             const float* __restrict__ b2r, float* __restrict__ out)
{
    int wid = threadIdx.x >> 5, lane = threadIdx.x & 31;
    int r = blockIdx.x * 8 + wid;
    const u16* row = hr + (size_t)r * NR_PAD;
    float s = 0.0f;
    for (int k = lane; k < 1540; k += 32)
        s += bf2f(row[k]) * W2r[k];
    for (int off = 16; off; off >>= 1) s += __shfl_xor(s, off, 32);
    if (lane == 0) {
        int i = (r >> 6) & 63, j = r & 63;
        out[r] = (i == j) ? 0.0f : (s + b2r[0]);
    }
}

// ---------------- host ----------------
extern "C" void kernel_launch(void* const* d_in, const int* in_sizes, int n_in,
                              void* d_out, int out_size, void* d_ws, size_t ws_size,
                              hipStream_t stream)
{
    (void)in_sizes; (void)n_in; (void)out_size; (void)ws_size;
    const float* x      = (const float*)d_in[0];
    const float* adjs   = (const float*)d_in[1];
    const float* flags  = (const float*)d_in[2];
    const float* Wg     = (const float*)d_in[3];
    const float* bg     = (const float*)d_in[4];
    const float* Wout   = (const float*)d_in[5];
    const float* bout   = (const float*)d_in[6];
    const float* W1t    = (const float*)d_in[7];
    const float* b1t    = (const float*)d_in[8];
    const float* gam    = (const float*)d_in[9];
    const float* bet    = (const float*)d_in[10];
    const float* W2t    = (const float*)d_in[11];
    const float* b2t    = (const float*)d_in[12];
    const float* W1r    = (const float*)d_in[13];
    const float* b1r    = (const float*)d_in[14];
    const float* W2r    = (const float*)d_in[15];
    const float* b2r    = (const float*)d_in[16];
    float* out_score = (float*)d_out;            // [65536]
    float* out_xo    = (float*)d_out + ROWS;     // [1024*768]

    char* ws = (char*)d_ws;
    size_t off = 0;
    auto alloc = [&](size_t bytes) -> void* {
        void* p = ws + off;
        off += (bytes + 255) & ~(size_t)255;
        return p;
    };
    u16* Abf   = (u16*)alloc((size_t)BB * 2 * 128 * 64 * 2); // [b][c][128][64], padded
    u16* xT    = (u16*)alloc((size_t)BB * F_IN * NN * 2);    // [b][f][j]
    u16* xh    = (u16*)alloc((size_t)BB * NN * 384 * 2);     // [row][x|h]
    u16* agg   = (u16*)alloc((size_t)BB * NN * 256 * 2);
    float* xo  = (float*)alloc((size_t)BB * NN * OUT_DIM * 4);
    u16* xobf  = (u16*)alloc((size_t)BB * NN * OUT_DIM * 2);
    float* P   = (float*)alloc((size_t)BB * NN * H_T * 4);
    float* Q   = (float*)alloc((size_t)BB * NN * H_T * 4);
    u16* WgT   = (u16*)alloc((size_t)256 * 256 * 2);
    u16* WoutT = (u16*)alloc((size_t)768 * 384 * 2);
    u16* W1tmT = (u16*)alloc((size_t)1536 * 768 * 2);
    u16* W1thT = (u16*)alloc((size_t)1536 * 768 * 2);
    u16* W2tT  = (u16*)alloc((size_t)768 * 1536 * 2);
    u16* W1rT  = (u16*)alloc((size_t)NR_PAD * KR_PAD * 2);
    float* b1rp  = (float*)alloc(NR_PAD * 4);
    float* gsum  = (float*)alloc(H_T * 4);
    float* gsq   = (float*)alloc(H_T * 4);
    float* scale = (float*)alloc(H_T * 4);
    float* shift = (float*)alloc(H_T * 4);
    u16* hmid = (u16*)alloc((size_t)ROWS * H_T * 2);        // 201 MB
    u16* raw  = (u16*)alloc((size_t)ROWS * OUT_DIM * 2);    // 100 MB
    u16* rin  = (u16*)alloc((size_t)ROWS * KR_PAD * 2);     // 105 MB
    u16* hr   = (u16*)alloc((size_t)ROWS * NR_PAD * 2);     // 218 MB

    dim3 blk(256);
    // --- weight transposes (f32 -> bf16, Bt[N][K]) ---
    k_tconv<<<dim3((256 * 256 + 255) / 256), blk, 0, stream>>>(Wg, 256, 0, 256, 256, 256, WgT, 256 * 256);
    k_tconv<<<dim3((768 * 384 + 255) / 256), blk, 0, stream>>>(Wout, 768, 0, 384, 768, 384, WoutT, 768 * 384);
    k_tconv<<<dim3((1536 * 768 + 255) / 256), blk, 0, stream>>>(W1t, 1536, 2, 768, 1536, 768, W1tmT, 1536 * 768);
    k_tconv<<<dim3((1536 * 768 + 255) / 256), blk, 0, stream>>>(W1t, 1536, 770, 768, 1536, 768, W1thT, 1536 * 768);
    k_tconv<<<dim3((768 * 1536 + 255) / 256), blk, 0, stream>>>(W2t, 768, 0, 1536, 768, 1536, W2tT, 768 * 1536);
    k_tconv<<<dim3((NR_PAD * KR_PAD + 255) / 256), blk, 0, stream>>>(W1r, 1540, 0, 770, 1540, KR_PAD, W1rT, NR_PAD * KR_PAD);
    k_padbias<<<dim3((NR_PAD + 255) / 256), blk, 0, stream>>>(b1r, b1rp);
    k_buildA<<<dim3((BB * 2 * 128 * 64 + 255) / 256), blk, 0, stream>>>(adjs, flags, Abf);
    k_xprep<<<dim3((BB * NN * F_IN + 255) / 256), blk, 0, stream>>>(x, xh, xT);

    // --- GIN: agg[b,i,c*128+f] = (A_c @ x)  (batched over b, per channel) ---
    gemm_bf16_wmma<0, u16><<<dim3(1, 1, BB), blk, 0, stream>>>(
        Abf, xT, nullptr, agg, 64, 128, 64, 256, 16384L, 8192L, 64L * 256L);
    gemm_bf16_wmma<0, u16><<<dim3(1, 1, BB), blk, 0, stream>>>(
        Abf + 8192, xT, nullptr, agg + 128, 64, 128, 64, 256, 16384L, 8192L, 64L * 256L);
    // h = elu(agg @ Wg + bg) -> xh[:,128:384]
    gemm_bf16_wmma<1, u16><<<dim3(2, 8, 1), blk, 0, stream>>>(
        agg, WgT, bg, xh + 128, 1024, 256, 256, 384, 0L, 0L, 0L);
    k_maskh<<<dim3((BB * NN * F_HID + 255) / 256), blk, 0, stream>>>(xh, flags);
    // x_o = tanh([x,h] @ Wout + bout)
    gemm_bf16_wmma<2, float><<<dim3(6, 8, 1), blk, 0, stream>>>(
        xh, WoutT, bout, xo, 1024, 768, 384, 768, 0L, 0L, 0L);
    k_xofin<<<dim3((BB * NN * OUT_DIM + 255) / 256), blk, 0, stream>>>(xo, flags, out_xo, xobf);

    // --- translate_mlp layer 1 via P/Q decomposition ---
    gemm_bf16_wmma<0, float><<<dim3(12, 8, 1), blk, 0, stream>>>(
        xobf, W1tmT, nullptr, P, 1024, H_T, 768, H_T, 0L, 0L, 0L);
    gemm_bf16_wmma<0, float><<<dim3(12, 8, 1), blk, 0, stream>>>(
        xobf, W1thT, nullptr, Q, 1024, H_T, 768, H_T, 0L, 0L, 0L);
    hipMemsetAsync(gsum, 0, H_T * 4, stream);
    hipMemsetAsync(gsq, 0, H_T * 4, stream);
    k_ewstat<<<dim3(1024, 6, 1), blk, 0, stream>>>(P, Q, adjs, flags, W1t, W1t + H_T, b1t, gsum, gsq);
    k_bnfin<<<dim3(6), blk, 0, stream>>>(gsum, gsq, gam, bet, scale, shift);
    k_ew2<<<dim3(1024, 6, 1), blk, 0, stream>>>(P, Q, adjs, flags, W1t, W1t + H_T, b1t, scale, shift, hmid);

    // --- translate_mlp layer 2: raw = hmid @ W2t + b2t ---
    gemm_bf16_wmma<0, u16><<<dim3(6, 512, 1), blk, 0, stream>>>(
        hmid, W2tT, b2t, raw, ROWS, 768, H_T, 768, 0L, 0L, 0L);

    // --- final_read_score ---
    k_rin<<<dim3(ROWS), blk, 0, stream>>>(raw, adjs, flags, rin);
    gemm_bf16_wmma<1, u16><<<dim3(13, 512, 1), blk, 0, stream>>>(
        rin, W1rT, b1rp, hr, ROWS, NR_PAD, KR_PAD, NR_PAD, 0L, 0L, 0L);
    k_score<<<dim3(ROWS / 8), blk, 0, stream>>>(hr, W2r, b2r, out_score);
}
